// NormAttention_30064771072520
// MI455X (gfx1250) — matrix operand order repaired and checked by reading, not compiled, not run on hardware
//
#include <hip/hip_runtime.h>
#include <hip/hip_bf16.h>

#define B_   4
#define N_   2048
#define C_   1024
#define H_   16
#define D_   64
#define BN   8192
#define EPSF 1e-6f

typedef __bf16 bf16_t;
typedef __attribute__((ext_vector_type(8)))  __bf16 v8bf;
typedef __attribute__((ext_vector_type(16))) __bf16 v16bf;
typedef __attribute__((ext_vector_type(8)))  float  v8f;
typedef int v4i_ __attribute__((vector_size(4 * sizeof(int))));

union Frag16 { v16bf v; v8bf h[2]; };

// A-matrix (16-bit, 16xK) per-lane fragment: elems 0..7 at base, elems 8..15 at base+16
static __device__ inline v16bf load_a_frag(const bf16_t* base) {
  Frag16 f;
  f.h[0] = *(const v8bf*)(base);
  f.h[1] = *(const v8bf*)(base + 16);
  return f.v;
}
// B-matrix fragment: elems 0..7 at base, elems 8..15 at base+16
static __device__ inline v16bf load_b_frag(const bf16_t* base) {
  Frag16 f;
  f.h[0] = *(const v8bf*)(base);
  f.h[1] = *(const v8bf*)(base + 16);
  return f.v;
}

#define WMMA_BF16(a, b, c) \
  __builtin_amdgcn_wmma_f32_16x16x32_bf16(false, (a), false, (b), (short)0, (c), false, false)

// --- async global->LDS staging (gfx1250, ASYNCcnt-tracked) ---
#define AS1_B128(p) ((__attribute__((address_space(1))) v4i_*)(uintptr_t)(p))
#define AS3_B128(p) ((__attribute__((address_space(3))) v4i_*)(void*)(p))
#define ASYNC_B128(g, l) \
  __builtin_amdgcn_global_load_async_to_lds_b128(AS1_B128(g), AS3_B128(l), 0, 0)

static __device__ inline void wait_async_le2() { asm volatile("s_wait_asynccnt 0x2" ::: "memory"); }
static __device__ inline void wait_async_le4() { asm volatile("s_wait_asynccnt 0x4" ::: "memory"); }
static __device__ inline void wait_async_0()  { asm volatile("s_wait_asynccnt 0x0" ::: "memory"); }

// ---------------- conversion kernels ----------------

__global__ __launch_bounds__(256) void k_cvt_x(const float* __restrict__ x,
                                               bf16_t* __restrict__ xb, int n) {
  int i = blockIdx.x * 256 + threadIdx.x;
  if (i < n) xb[i] = (bf16_t)x[i];
}

// w: [K][Ncols] row-major f32  ->  wT: [Ncols][K] bf16
__global__ __launch_bounds__(256) void k_transpose_w(const float* __restrict__ w,
                                                     bf16_t* __restrict__ wT,
                                                     int K, int Ncols) {
  int n = blockIdx.x * 16 + (threadIdx.x & 15);
  int k = blockIdx.y * 16 + (threadIdx.x >> 4);
  if (n < Ncols && k < K) wT[(size_t)n * K + k] = (bf16_t)w[(size_t)k * Ncols + n];
}

// ---------------- fused QKV GEMM + RMSNorm + RoPE ----------------
// grid (48, 64), block 128 (4 waves). Block tile 128x64; wave tile 32x64.
// B tile (64 cols x 32 k) staged into LDS by async DMA, double-buffered.
// K-loop kept rolled (pointer-XOR double buffer) so accumulators/frags stay
// register-pinned: loads (VMEM/DS) overwrite frags hazard-free, no VALU movs.
__global__ __launch_bounds__(128)
void k_qkv(const bf16_t* __restrict__ xb, const bf16_t* __restrict__ wT,
           const float* __restrict__ qn_w, const float* __restrict__ kn_w,
           const float* __restrict__ fcos, const float* __restrict__ fsin,
           bf16_t* __restrict__ qb, bf16_t* __restrict__ kb, bf16_t* __restrict__ vt) {
  __shared__ __align__(16) bf16_t Bt[2][64][32];

  const int lane = threadIdx.x & 31;
  const int wv   = threadIdx.x >> 5;
  const int lr   = lane & 15;
  const int hi   = lane >> 4;
  const int m0   = blockIdx.y * 128 + wv * 32;
  const int n0   = blockIdx.x * 64;

  // async staging map: thread -> (col, 8-elem k segment); covers k {s..s+7, s+16..s+23}
  const int scol = threadIdx.x & 63;
  const int sseg = (threadIdx.x >> 6) * 8;
  const bf16_t* srow = wT + (size_t)(n0 + scol) * C_ + sseg;

  v8f acc[2][4] = {};
  const bf16_t* arow0 = xb + (size_t)(m0 + lr) * C_ + hi * 8;
  const bf16_t* arow1 = arow0 + (size_t)16 * C_;

  bf16_t* const bt0 = &Bt[0][0][0];
  bf16_t* stg = bt0 + (size_t)(scol * 32 + sseg);   // this thread's staging slot, buf 0
  const bf16_t* frg = bt0 + (size_t)(lr * 32 + hi * 8);  // this lane's frag base, buf 0

  // prologue: stage chunk 0 into buffer 0
  ASYNC_B128(srow,      stg);
  ASYNC_B128(srow + 16, stg + 16);

  unsigned cofs = 0;                 // compute-buffer element offset (0 or 2048)
#pragma unroll 1
  for (int i = 0; i < C_ / 32; ++i) {
    const int kk = i * 32;
    const unsigned nofs = cofs ^ 2048;
    if (i + 1 < C_ / 32) {
      ASYNC_B128(srow + kk + 32, stg + nofs);
      ASYNC_B128(srow + kk + 48, stg + nofs + 16);
      wait_async_le2();              // in-order => chunk i has landed
    } else {
      wait_async_0();
    }
    __syncthreads();                 // chunk i visible to all waves
    __builtin_prefetch(arow0 + kk + 128, 0, 0);
    v16bf a0 = load_a_frag(arow0 + kk);
    v16bf a1 = load_a_frag(arow1 + kk);
#pragma unroll
    for (int nt = 0; nt < 4; ++nt) {
      v16bf b = load_b_frag(frg + cofs + nt * 16 * 32);
      acc[0][nt] = WMMA_BF16(a0, b, acc[0][nt]);
      acc[1][nt] = WMMA_BF16(a1, b, acc[1][nt]);
    }
    __syncthreads();                 // done reading buf before it is re-staged
    cofs = nofs;
  }

  const int which = n0 >> 10;        // 0=q 1=k 2=v
  const int head  = (n0 >> 6) & 15;

  if (which == 2) {
    // V: store transposed [B][H][D][N] (bf16) so PV B-frags are contiguous
#pragma unroll
    for (int mt = 0; mt < 2; ++mt)
#pragma unroll
      for (int r = 0; r < 8; ++r) {
        int t = m0 + mt * 16 + r + 8 * hi;
        int b = t >> 11, n = t & 2047;
#pragma unroll
        for (int nt = 0; nt < 4; ++nt) {
          int d = nt * 16 + lr;
          vt[((size_t)(b * H_ + head) * D_ + d) * N_ + n] = (bf16_t)acc[mt][nt][r];
        }
      }
  } else {
    float wn[4];
#pragma unroll
    for (int nt = 0; nt < 4; ++nt) {
      int d = nt * 16 + lr;
      wn[nt] = (which == 0) ? qn_w[d] : kn_w[d];
    }
    bf16_t* dst = (which == 0) ? qb : kb;
    const float sc = (which == 0) ? 0.125f : 1.0f;   // fold D^-1/2 into Q
#pragma unroll
    for (int mt = 0; mt < 2; ++mt)
#pragma unroll
      for (int r = 0; r < 8; ++r) {
        int t = m0 + mt * 16 + r + 8 * hi;
        int b = t >> 11, n = t & 2047;
        float ss = 0.f;
#pragma unroll
        for (int nt = 0; nt < 4; ++nt) { float v = acc[mt][nt][r]; ss += v * v; }
#pragma unroll
        for (int off = 1; off < 16; off <<= 1) ss += __shfl_xor(ss, off, 32);
        float inv = rsqrtf(ss * (1.0f / D_) + EPSF);
#pragma unroll
        for (int nt = 0; nt < 4; ++nt) {
          int d = nt * 16 + lr;
          float v = acc[mt][nt][r] * inv * wn[nt];
          float part = __shfl_xor(v, 1, 32);       // RoPE partner d^1 in lane^1
          float rot  = (d & 1) ? part : -part;
          float co = fcos[n * D_ + d], si = fsin[n * D_ + d];
          float o  = (v * co + rot * si) * sc;
          dst[((size_t)(b * H_ + head) * N_ + n) * D_ + d] = (bf16_t)o;
        }
      }
  }
}

// ---------------- flash attention ----------------
// grid (B*H=64, N/64=32), block 128 (4 waves, 16 q-rows each).
// K (32x64) and V (64x32) key-chunk tiles async-staged to LDS, double-buffered,
// shared by all 4 waves (4x global traffic cut vs per-wave loads).
__global__ __launch_bounds__(128)
void k_attn(const bf16_t* __restrict__ qb, const bf16_t* __restrict__ kb,
            const bf16_t* __restrict__ vt, bf16_t* __restrict__ ab) {
  __shared__ __align__(16) bf16_t Kt[2][32][64];
  __shared__ __align__(16) bf16_t Vl[2][64][32];
  __shared__ __align__(16) bf16_t Pl[4][16][32];   // per-wave P transpose staging

  const int tid  = threadIdx.x;
  const int lane = tid & 31;
  const int wv   = tid >> 5;
  const int lr   = lane & 15;
  const int hi   = lane >> 4;
  const int bh   = blockIdx.x;
  const int q0   = blockIdx.y * 64 + wv * 16;
  const int b    = bh >> 4, h = bh & 15;

  const bf16_t* qrow  = qb + ((size_t)bh * N_ + q0 + lr) * D_;
  const v16bf qa0 = load_a_frag(qrow + hi * 8);        // d 0..31
  const v16bf qa1 = load_a_frag(qrow + 32 + hi * 8);   // d 32..63

  const bf16_t* kbhp = kb + (size_t)bh * N_ * D_;
  const bf16_t* vbhp = vt + (size_t)bh * D_ * N_;

  // staging maps
  const int kjl = tid & 31;             // K: key row in chunk
  const int kds = (tid >> 5) * 16;      // K: d segment (0/16/32/48)
  const int vdr = tid & 63;             // V: d row
  const int vjs = (tid >> 6) * 16;      // V: key segment (0/16)
  bf16_t* const kt0 = &Kt[0][0][0];
  bf16_t* const vl0 = &Vl[0][0][0];
  bf16_t* kstg = kt0 + (size_t)(kjl * 64 + kds);
  bf16_t* vstg = vl0 + (size_t)(vdr * 32 + vjs);

  float m[8], l[8], alpha[8];
  v8f o[4] = {};
#pragma unroll
  for (int r = 0; r < 8; ++r) { m[r] = -1e30f; l[r] = 0.f; }
  const v8f vzero = {};

  // prologue: stage key-chunk 0 into buffer 0
  {
    const bf16_t* ks = kbhp + (size_t)kjl * D_ + kds;
    ASYNC_B128(ks,     kstg);
    ASYNC_B128(ks + 8, kstg + 8);
    const bf16_t* vs = vbhp + (size_t)vdr * N_ + vjs;
    ASYNC_B128(vs,     vstg);
    ASYNC_B128(vs + 8, vstg + 8);
  }

  unsigned cofs = 0;                   // compute-buffer element offset (0 or 2048)
#pragma unroll 1
  for (int j0 = 0; j0 < N_; j0 += 32) {
    const unsigned nofs = cofs ^ 2048;
    if (j0 + 32 < N_) {
      const bf16_t* ks = kbhp + (size_t)(j0 + 32 + kjl) * D_ + kds;
      ASYNC_B128(ks,     kstg + nofs);
      ASYNC_B128(ks + 8, kstg + nofs + 8);
      const bf16_t* vs = vbhp + (size_t)vdr * N_ + j0 + 32 + vjs;
      ASYNC_B128(vs,     vstg + nofs);
      ASYNC_B128(vs + 8, vstg + nofs + 8);
      wait_async_le4();                // in-order => chunk j0 has landed
    } else {
      wait_async_0();
    }
    __syncthreads();                   // chunk j0 visible to all waves

    // S = Q K^T (scale folded into Q); K B-frags from LDS
    v8f s[2];
#pragma unroll
    for (int jt = 0; jt < 2; ++jt) {
      const bf16_t* kr = kt0 + cofs + (size_t)(jt * 16 + lr) * 64 + hi * 8;
      v16bf b0 = load_b_frag(kr);        // d 0..31 slice
      v16bf b1 = load_b_frag(kr + 32);   // d 32..63 slice
      s[jt] = WMMA_BF16(qa0, b0, vzero);
      s[jt] = WMMA_BF16(qa1, b1, s[jt]);
    }
#pragma unroll
    for (int r = 0; r < 8; ++r) {
      float mx = fmaxf(s[0][r], s[1][r]);
#pragma unroll
      for (int off = 1; off < 16; off <<= 1) mx = fmaxf(mx, __shfl_xor(mx, off, 32));
      float nm = fmaxf(m[r], mx);
      float a  = __expf(m[r] - nm);
      float p0 = __expf(s[0][r] - nm);
      float p1 = __expf(s[1][r] - nm);
      float rs = p0 + p1;
#pragma unroll
      for (int off = 1; off < 16; off <<= 1) rs += __shfl_xor(rs, off, 32);
      l[r] = l[r] * a + rs;
      m[r] = nm;
      alpha[r] = a;
      Pl[wv][r + 8 * hi][lr]      = (bf16_t)p0;
      Pl[wv][r + 8 * hi][16 + lr] = (bf16_t)p1;
    }
#pragma unroll
    for (int nt = 0; nt < 4; ++nt)
#pragma unroll
      for (int r = 0; r < 8; ++r) o[nt][r] *= alpha[r];

    // per-wave DS ops are in-order: Pl stores above are seen by this wave's loads
    v16bf pa = load_a_frag(&Pl[wv][lr][hi * 8]);
#pragma unroll
    for (int nt = 0; nt < 4; ++nt) {
      v16bf vb = load_b_frag(vl0 + cofs + (size_t)(nt * 16 + lr) * 32 + hi * 8);
      o[nt] = WMMA_BF16(pa, vb, o[nt]);
    }
    __syncthreads();                   // all waves done reading buf before restage
    cofs = nofs;
  }

#pragma unroll
  for (int r = 0; r < 8; ++r) {
    int row = q0 + r + 8 * hi;
    float invl = 1.0f / l[r];
#pragma unroll
    for (int nt = 0; nt < 4; ++nt) {
      int d = nt * 16 + lr;
      ab[((size_t)(b * N_ + row)) * C_ + h * D_ + d] = (bf16_t)(o[nt][r] * invl);
    }
  }
}

// ---------------- output projection GEMM + bias ----------------
// grid (16, 64), block 128 (4 waves), block tile 128x64, async-staged B, f32 out.
__global__ __launch_bounds__(128)
void k_proj(const bf16_t* __restrict__ ab, const bf16_t* __restrict__ wT,
            const float* __restrict__ bias, float* __restrict__ out) {
  __shared__ __align__(16) bf16_t Bt[2][64][32];

  const int lane = threadIdx.x & 31;
  const int wv   = threadIdx.x >> 5;
  const int lr   = lane & 15;
  const int hi   = lane >> 4;
  const int m0   = blockIdx.y * 128 + wv * 32;
  const int n0   = blockIdx.x * 64;

  const int scol = threadIdx.x & 63;
  const int sseg = (threadIdx.x >> 6) * 8;
  const bf16_t* srow = wT + (size_t)(n0 + scol) * C_ + sseg;

  v8f acc[2][4] = {};
  const bf16_t* arow0 = ab + (size_t)(m0 + lr) * C_ + hi * 8;
  const bf16_t* arow1 = arow0 + (size_t)16 * C_;

  bf16_t* const bt0 = &Bt[0][0][0];
  bf16_t* stg = bt0 + (size_t)(scol * 32 + sseg);
  const bf16_t* frg = bt0 + (size_t)(lr * 32 + hi * 8);

  ASYNC_B128(srow,      stg);
  ASYNC_B128(srow + 16, stg + 16);

  unsigned cofs = 0;
#pragma unroll 1
  for (int i = 0; i < C_ / 32; ++i) {
    const int kk = i * 32;
    const unsigned nofs = cofs ^ 2048;
    if (i + 1 < C_ / 32) {
      ASYNC_B128(srow + kk + 32, stg + nofs);
      ASYNC_B128(srow + kk + 48, stg + nofs + 16);
      wait_async_le2();
    } else {
      wait_async_0();
    }
    __syncthreads();
    __builtin_prefetch(arow0 + kk + 128, 0, 0);
    v16bf a0 = load_a_frag(arow0 + kk);
    v16bf a1 = load_a_frag(arow1 + kk);
#pragma unroll
    for (int nt = 0; nt < 4; ++nt) {
      v16bf b = load_b_frag(frg + cofs + nt * 16 * 32);
      acc[0][nt] = WMMA_BF16(a0, b, acc[0][nt]);
      acc[1][nt] = WMMA_BF16(a1, b, acc[1][nt]);
    }
    __syncthreads();
    cofs = nofs;
  }
#pragma unroll
  for (int mt = 0; mt < 2; ++mt)
#pragma unroll
    for (int r = 0; r < 8; ++r) {
      int t = m0 + mt * 16 + r + 8 * hi;
#pragma unroll
      for (int nt = 0; nt < 4; ++nt) {
        int col = n0 + nt * 16 + lr;
        out[(size_t)t * C_ + col] = acc[mt][nt][r] + bias[col];
      }
    }
}

// ---------------- launcher ----------------
extern "C" void kernel_launch(void* const* d_in, const int* in_sizes, int n_in,
                              void* d_out, int out_size, void* d_ws, size_t ws_size,
                              hipStream_t stream) {
  const float* x      = (const float*)d_in[0];
  const float* w_qkv  = (const float*)d_in[1];
  const float* w_proj = (const float*)d_in[2];
  const float* b_proj = (const float*)d_in[3];
  const float* qn_w   = (const float*)d_in[4];
  const float* kn_w   = (const float*)d_in[5];
  const float* fcos   = (const float*)d_in[6];
  const float* fsin   = (const float*)d_in[7];

  char* ws = (char*)d_ws;
  size_t off = 0;
  bf16_t* xb     = (bf16_t*)(ws + off); off += (size_t)BN * C_ * 2;      // 16 MB
  bf16_t* wqkvT  = (bf16_t*)(ws + off); off += (size_t)3 * C_ * C_ * 2;  //  6 MB
  bf16_t* wprojT = (bf16_t*)(ws + off); off += (size_t)C_ * C_ * 2;      //  2 MB
  bf16_t* qb     = (bf16_t*)(ws + off); off += (size_t)BN * C_ * 2;      // 16 MB
  bf16_t* kbuf   = (bf16_t*)(ws + off); off += (size_t)BN * C_ * 2;      // 16 MB
  bf16_t* vt     = (bf16_t*)(ws + off); off += (size_t)BN * C_ * 2;      // 16 MB
  bf16_t* abuf   = (bf16_t*)(ws + off); off += (size_t)BN * C_ * 2;      // 16 MB

  k_cvt_x<<<(BN * C_) / 256, 256, 0, stream>>>(x, xb, BN * C_);
  k_transpose_w<<<dim3(3 * C_ / 16, C_ / 16), 256, 0, stream>>>(w_qkv, wqkvT, C_, 3 * C_);
  k_transpose_w<<<dim3(C_ / 16, C_ / 16), 256, 0, stream>>>(w_proj, wprojT, C_, C_);
  k_qkv<<<dim3(48, 64), 128, 0, stream>>>(xb, wqkvT, qn_w, kn_w, fcos, fsin, qb, kbuf, vt);
  k_attn<<<dim3(B_ * H_, N_ / 64), 128, 0, stream>>>(qb, kbuf, vt, abuf);
  k_proj<<<dim3(16, 64), 128, 0, stream>>>(abuf, wprojT, b_proj, (float*)d_out);
}
